// TarsBlock_15238543966336
// MI455X (gfx1250) — matrix operand order repaired and checked by reading, not compiled, not run on hardware
//
#include <hip/hip_runtime.h>
#include <hip/hip_bf16.h>

// ---------------------------------------------------------------------------
// Shapes (fixed by the reference)
// ---------------------------------------------------------------------------
#define BB   2
#define LL   1024
#define DD   768
#define TT   (BB*LL)          // 2048 token rows
#define HH   12
#define PP   64
#define NNR  64
#define OMD  32
#define MEMD 384
#define D2   1536

typedef __attribute__((ext_vector_type(16))) __bf16 v16bf;
typedef __attribute__((ext_vector_type(8)))  __bf16 v8bf;
typedef __attribute__((ext_vector_type(8)))  float  v8f;
typedef __attribute__((ext_vector_type(4)))  unsigned int u32x4;
typedef __attribute__((ext_vector_type(8)))  int i32x8;
typedef __attribute__((ext_vector_type(4)))  int i32x4;

#define DEV __device__ __forceinline__

#if __has_builtin(__builtin_amdgcn_tensor_load_to_lds)
#define HAVE_TDM 1
#else
#define HAVE_TDM 0
#endif

DEV float sig_(float x){ return 1.0f/(1.0f+expf(-x)); }
DEV float silu_(float x){ return x*sig_(x); }

DEV unsigned short f2bf(float f){
  union { float f; unsigned int u; } c; c.f = f;
  unsigned int u = c.u;
  u += 0x7fffu + ((u >> 16) & 1u);       // round-to-nearest-even
  return (unsigned short)(u >> 16);
}

DEV float blk_reduce(float v, float* red){
  int tid = threadIdx.x;
  red[tid] = v; __syncthreads();
  for (int off = (int)blockDim.x >> 1; off > 0; off >>= 1){
    if (tid < off) red[tid] += red[tid+off];
    __syncthreads();
  }
  float r = red[0]; __syncthreads();
  return r;
}

// ---------------------------------------------------------------------------
// Weight convert + transpose (+ zero pad):  src[K,Nout] f32 -> dst[Npad,K] bf16
// ---------------------------------------------------------------------------
__global__ void k_wconv_t(const float* __restrict__ src, unsigned short* __restrict__ dst,
                          int K, int Nout, int Npad){
  long idx = (long)blockIdx.x*blockDim.x + threadIdx.x;
  if (idx >= (long)Npad*K) return;
  int n = (int)(idx / K), kk = (int)(idx % K);
  float v = (n < Nout) ? src[(long)kk*Nout + n] : 0.0f;
  dst[idx] = f2bf(v);
}

// Activation convert: f32 -> bf16 (same layout)
__global__ void k_cvt_bf16(const float* __restrict__ src, unsigned short* __restrict__ dst, long n){
  long idx = (long)blockIdx.x*blockDim.x + threadIdx.x;
  if (idx < n) dst[idx] = f2bf(src[idx]);
}

// ---------------------------------------------------------------------------
// Direct-load WMMA GEMM (odd shapes): one wave = one 16x16 tile.
// ---------------------------------------------------------------------------
__global__ void __launch_bounds__(32)
k_gemm_wmma(const __bf16* __restrict__ A, const __bf16* __restrict__ Wt,
            float* __restrict__ C, int K, int Nout){
  const int lane = threadIdx.x;
  const int tm = blockIdx.x, tn = blockIdx.y;
  const int half = lane >> 4;
  const int l15  = lane & 15;

  const __bf16* pa = A + (size_t)(tm*16 + l15)*K + half*8;     // A 16x32 frag (ISA 7.12.2)
  const __bf16* pb = Wt + (size_t)(tn*16 + l15)*K + half*8;    // B 32x16 frag

  v8f acc = {};
  for (int k0 = 0; k0 < K; k0 += 32){
    v8bf a0 = *(const v8bf*)(pa + k0);
    v8bf a1 = *(const v8bf*)(pa + k0 + 16);
    v8bf b0 = *(const v8bf*)(pb + k0);
    v8bf b1 = *(const v8bf*)(pb + k0 + 16);
    if (k0 + 32 < K){
      __builtin_prefetch(pa + k0 + 32, 0, 1);   // global_prefetch_b8
      __builtin_prefetch(pb + k0 + 32, 0, 1);
    }
    v16bf av, bv;
#pragma unroll
    for (int i = 0; i < 8; ++i){
      av[i] = a0[i]; av[8+i] = a1[i];
      bv[i] = b0[i]; bv[8+i] = b1[i];
    }
    acc = __builtin_amdgcn_wmma_f32_16x16x32_bf16(
        false, av, false, bv, (short)0, acc, false, false);
  }
  const int col = tn*16 + l15;
  if (col < Nout){
#pragma unroll
    for (int r = 0; r < 8; ++r){
      int row = tm*16 + half*8 + r;
      C[(size_t)row*Nout + col] = acc[r];
    }
  }
}

// ---------------------------------------------------------------------------
// TDM-fed LDS-tiled WMMA GEMM: 128 threads (4 waves), 64x64 macro-tile,
// K-step 64, double-buffered tensor_load_to_lds with LDS pad (row pitch 144B
// = 72 bf16 -> ds_load_b128 fragment reads are bank-conflict free).
// ---------------------------------------------------------------------------
#define TPITCH 72                 // bf16 elements per LDS tile row (64 + 8 pad)
#define TILE_USH 4608             // ushorts per 64-row tile (64*72)
#define TILE_BYTES 9216

#if HAVE_TDM
DEV void tdm_load_tile64(unsigned lds_off, const void* gp, unsigned stride_elems){
  unsigned long long ga = (unsigned long long)gp;
  u32x4 g0;
  g0[0] = 1u;                                              // count=1 (valid D#)
  g0[1] = lds_off;                                         // lds_addr (bytes)
  g0[2] = (unsigned)ga;                                    // global_addr[31:0]
  g0[3] = ((unsigned)(ga >> 32) & 0x01ffffffu) | (2u << 30); // addr[56:32] | type=2
  i32x8 g1;
  // data_size=1(2B) | pad_enable | pad_interval=4(32 DW =128B) | pad_amount=3(4 DW =16B)
  g1[0] = (int)((1u << 16) | (1u << 20) | (4u << 22) | (3u << 25));
  g1[1] = (int)(64u << 16);            // tensor_dim0 lo = 64
  g1[2] = (int)(64u << 16);            // tensor_dim0 hi=0 | tensor_dim1 lo = 64
  g1[3] = (int)(64u << 16);            // tensor_dim1 hi=0 | tile_dim0 = 64
  g1[4] = 64;                          // tile_dim1 = 64 | tile_dim2 = 0
  g1[5] = (int)stride_elems;           // tensor_dim0_stride[31:0] (row pitch, elems)
  g1[6] = 0;                           // stride[47:32] | tensor_dim1_stride lo
  g1[7] = 0;
  i32x4 z4 = {0,0,0,0};
#if __clang_major__ >= 23
  i32x8 z8 = {0,0,0,0,0,0,0,0};
  __builtin_amdgcn_tensor_load_to_lds(g0, g1, z4, z4, z8, 0);
#else
  __builtin_amdgcn_tensor_load_to_lds(g0, g1, z4, z4, 0);
#endif
}
#else
DEV void copy_tile64(unsigned short* lds, const unsigned short* g, int stride, int tid){
  // 64 rows x 64 elems, LDS pitch TPITCH; 8-elem (16B) chunks
  for (int idx = tid; idx < 64*8; idx += 128){
    int r = idx >> 3, c8 = (idx & 7)*8;
    *(v8bf*)(lds + r*TPITCH + c8) = *(const v8bf*)(g + (size_t)r*stride + c8);
  }
}
#endif

__global__ void __launch_bounds__(128)
k_gemm_wmma_tdm(const __bf16* __restrict__ A, const __bf16* __restrict__ Wt,
                float* __restrict__ C, int K, int Nout){
  __shared__ __align__(16) unsigned short smem[4*TILE_USH];  // [buf][A|B] tiles
  const int tid = threadIdx.x;
  const int wave = tid >> 5, lane = tid & 31;
  const int half = lane >> 4, l15 = lane & 15;
  const int row0 = blockIdx.x*64;
  const int col0 = blockIdx.y*64;
  const int nk = K / 64;

  v8f acc[4] = {};

#if HAVE_TDM
  if (wave == 0){
    tdm_load_tile64(0u*TILE_BYTES, A  + (size_t)row0*K, (unsigned)K);
    tdm_load_tile64(1u*TILE_BYTES, Wt + (size_t)col0*K, (unsigned)K);
  }
#endif
  for (int kc = 0; kc < nk; ++kc){
    const int buf = kc & 1;
#if HAVE_TDM
    if (wave == 0){
      if (kc + 1 < nk){
        const int nb = 1 - buf;
        tdm_load_tile64((unsigned)(nb*2+0)*TILE_BYTES, A  + (size_t)row0*K + (kc+1)*64, (unsigned)K);
        tdm_load_tile64((unsigned)(nb*2+1)*TILE_BYTES, Wt + (size_t)col0*K + (kc+1)*64, (unsigned)K);
        __builtin_amdgcn_s_wait_tensorcnt(2);   // current pair done; next may fly
      } else {
        __builtin_amdgcn_s_wait_tensorcnt(0);
      }
    }
    __syncthreads();
#else
    copy_tile64(smem + (buf*2+0)*TILE_USH, (const unsigned short*)A  + (size_t)row0*K + kc*64, K, tid);
    copy_tile64(smem + (buf*2+1)*TILE_USH, (const unsigned short*)Wt + (size_t)col0*K + kc*64, K, tid);
    __syncthreads();
#endif
    const __bf16* At = (const __bf16*)(smem + (buf*2+0)*TILE_USH);
    const __bf16* Bt = (const __bf16*)(smem + (buf*2+1)*TILE_USH);
#pragma unroll
    for (int kt = 0; kt < 2; ++kt){
      const __bf16* pa = At + (size_t)(wave*16 + l15)*TPITCH + kt*32 + half*8;
      v8bf a0 = *(const v8bf*)pa;
      v8bf a1 = *(const v8bf*)(pa + 16);
      v16bf av;
#pragma unroll
      for (int i = 0; i < 8; ++i){ av[i] = a0[i]; av[8+i] = a1[i]; }
#pragma unroll
      for (int nt = 0; nt < 4; ++nt){
        const __bf16* pb = Bt + (size_t)(nt*16 + l15)*TPITCH + kt*32 + half*8;
        v8bf b0 = *(const v8bf*)pb;
        v8bf b1 = *(const v8bf*)(pb + 16);
        v16bf bv;
#pragma unroll
        for (int i = 0; i < 8; ++i){ bv[i] = b0[i]; bv[8+i] = b1[i]; }
        acc[nt] = __builtin_amdgcn_wmma_f32_16x16x32_bf16(
            false, av, false, bv, (short)0, acc[nt], false, false);
      }
    }
    __syncthreads();
  }
  const int rbase = row0 + wave*16 + half*8;
#pragma unroll
  for (int nt = 0; nt < 4; ++nt){
    const int col = col0 + nt*16 + l15;
#pragma unroll
    for (int r = 0; r < 8; ++r)
      C[(size_t)(rbase + r)*Nout + col] = acc[nt][r];
  }
}

// ---------------------------------------------------------------------------
// Elementwise / small kernels
// ---------------------------------------------------------------------------
__global__ void k_ew_pred(const float* __restrict__ x, const float* __restrict__ pred,
                          float* __restrict__ out, long n){
  long i = (long)blockIdx.x*blockDim.x + threadIdx.x;
  if (i < n) out[i] = 1.05f*x[i] - 0.05f*pred[i];
}

__global__ void __launch_bounds__(256)
k_ln(const float* __restrict__ in, const float* __restrict__ g,
     const float* __restrict__ b, float* __restrict__ out){
  __shared__ float red[256];
  const int row = blockIdx.x;
  const float* xr = in + (size_t)row*DD;
  float s = 0.f;
  for (int d = threadIdx.x; d < DD; d += 256) s += xr[d];
  float mean = blk_reduce(s, red) / DD;
  float v = 0.f;
  for (int d = threadIdx.x; d < DD; d += 256){ float t = xr[d]-mean; v += t*t; }
  float var = blk_reduce(v, red) / DD;
  float inv = rsqrtf(var + 1e-5f);
  for (int d = threadIdx.x; d < DD; d += 256)
    out[(size_t)row*DD + d] = (xr[d]-mean)*inv*g[d] + b[d];
}

__global__ void k_shift_mix(const float* __restrict__ xn, const float* __restrict__ x_prev,
                            const float* __restrict__ mu, float* __restrict__ xm, long n){
  long i = (long)blockIdx.x*blockDim.x + threadIdx.x;
  if (i >= n) return;
  int d = (int)(i % DD); long t = i / DD; int b = (int)(t / LL); int l = (int)(t % LL);
  float cur  = xn[i];
  float prev = (l == 0) ? x_prev[(size_t)b*DD + d] : xn[i - DD];
  xm[i] = mu[d]*cur + (1.0f-mu[d])*prev;
}

__global__ void k_conv_silu(const float* __restrict__ xz, const float* __restrict__ cst,
                            const float* __restrict__ cw, const float* __restrict__ cb,
                            float* __restrict__ xc, long n){
  long i = (long)blockIdx.x*blockDim.x + threadIdx.x;
  if (i >= n) return;
  int d = (int)(i % DD); long t = i / DD; int b = (int)(t / LL); int l = (int)(t % LL);
  float acc = cb[d];
#pragma unroll
  for (int k = 0; k < 4; ++k){
    int p = l + k;
    float pv = (p < 3) ? cst[((size_t)b*DD + d)*3 + p]
                       : xz[((size_t)b*LL + (p-3))*D2 + d];   // xc0 = xz[:, :D]
    acc += pv * cw[d*4 + k];
  }
  xc[i] = silu_(acc);
}

__global__ void k_ew_dt(const float* __restrict__ dtp, const float* __restrict__ dt_bias,
                        const float* __restrict__ A_log, float* __restrict__ dt,
                        float* __restrict__ dA, long n){
  long i = (long)blockIdx.x*blockDim.x + threadIdx.x;
  if (i >= n) return;
  int h = (int)(i % HH);
  float v = dtp[i] + dt_bias[h];
  float sp = (v > 20.0f) ? v : log1pf(expf(v));
  dt[i] = sp;
  dA[i] = expf(-expf(A_log[h]) * sp);
}

__global__ void k_ew_sigmoid(float* __restrict__ a, long n){
  long i = (long)blockIdx.x*blockDim.x + threadIdx.x;
  if (i < n) a[i] = sig_(a[i]);
}
__global__ void k_ew_expnegexp(float* __restrict__ a, long n){
  long i = (long)blockIdx.x*blockDim.x + threadIdx.x;
  if (i < n) a[i] = expf(-expf(a[i]));
}

__global__ void k_gfuse(const float* __restrict__ xm, const float* __restrict__ wf,
                        float* __restrict__ g, int T){
  int t = blockIdx.x*blockDim.x + threadIdx.x;
  if (t >= T) return;
  float s = 0.f;
  const float* xr = xm + (size_t)t*DD;
  for (int d = 0; d < DD; ++d) s += xr[d]*wf[d];
  g[t] = sig_(s);
}

// ---------------------------------------------------------------------------
// SSD scan: one block per (b,h); state P x N = 64x64 in registers (16/lane)
// ---------------------------------------------------------------------------
__global__ void __launch_bounds__(256)
k_ssd_scan(const float* __restrict__ dA, const float* __restrict__ dt,
           const float* __restrict__ xc, const float* __restrict__ Bm,
           const float* __restrict__ Cm, const float* __restrict__ s0,
           float* __restrict__ ys){
  const int bh = blockIdx.x; const int b = bh / HH; const int h = bh % HH;
  const int tid = threadIdx.x;
  const int p = tid >> 2; const int ng = tid & 3;
  __shared__ float Bs[NNR], Cs[NNR], us[PP];
  __shared__ float part[256];
  float s[16];
  const float* sp = s0 + (((size_t)b*HH + h)*PP + p)*NNR + ng*16;
#pragma unroll
  for (int i = 0; i < 16; ++i) s[i] = sp[i];
  for (int t = 0; t < LL; ++t){
    const size_t row = (size_t)b*LL + t;
    if (tid < NNR){
      Bs[tid] = Bm[row*NNR + tid];
      Cs[tid] = Cm[row*NNR + tid];
      us[tid] = dt[row*HH + h] * xc[row*DD + h*PP + tid];
    }
    __syncthreads();
    const float a = dA[row*HH + h];
    const float u = us[p];
    float acc = 0.f;
#pragma unroll
    for (int i = 0; i < 16; ++i){
      int n = ng*16 + i;
      s[i] = a*s[i] + u*Bs[n];
      acc += s[i]*Cs[n];
    }
    part[tid] = acc; __syncthreads();
    if (ng == 0)
      ys[row*DD + h*PP + p] = part[tid] + part[tid+1] + part[tid+2] + part[tid+3];
    __syncthreads();
  }
}

// ---------------------------------------------------------------------------
// RWKV scan: one block per batch; state N x N = 64x64 in registers
// ---------------------------------------------------------------------------
__global__ void __launch_bounds__(256)
k_rwkv_scan(const float* __restrict__ w, const float* __restrict__ k,
            const float* __restrict__ v, const float* __restrict__ r,
            const float* __restrict__ S0, float* __restrict__ os){
  const int b = blockIdx.x; const int tid = threadIdx.x;
  const int m = tid & 63; const int ng = tid >> 6;
  __shared__ float ws[NNR], ks[NNR], vs[NNR], rs[NNR];
  __shared__ float part[256];
  float S[16];
  const float* sp = S0 + ((size_t)b*NNR + ng*16)*NNR + m;
#pragma unroll
  for (int i = 0; i < 16; ++i) S[i] = sp[(size_t)i*NNR];
  for (int t = 0; t < LL; ++t){
    const size_t row = (size_t)b*LL + t;
    if (tid < NNR){
      ws[tid] = w[row*NNR + tid]; ks[tid] = k[row*NNR + tid];
      vs[tid] = v[row*NNR + tid]; rs[tid] = r[row*NNR + tid];
    }
    __syncthreads();
    const float vm = vs[m];
    float acc = 0.f;
#pragma unroll
    for (int i = 0; i < 16; ++i){
      int n = ng*16 + i;
      S[i] = ws[n]*S[i] + ks[n]*vm;
      acc += rs[n]*S[i];
    }
    part[(size_t)ng*64 + m] = acc; __syncthreads();
    if (ng == 0)
      os[row*NNR + m] = part[m] + part[64+m] + part[128+m] + part[192+m];
    __syncthreads();
  }
}

__global__ void k_combine(const float* __restrict__ g, const float* __restrict__ ys,
                          const float* __restrict__ xc, const float* __restrict__ yr,
                          const float* __restrict__ xz, const float* __restrict__ Dsk,
                          float* __restrict__ cp, long n){
  long i = (long)blockIdx.x*blockDim.x + threadIdx.x;
  if (i >= n) return;
  int d = (int)(i % DD); long t = i / DD;
  float yssd = ys[i] + Dsk[d >> 6]*xc[i];
  float gg = g[t];
  float zz = xz[t*D2 + DD + d];
  cp[i] = (gg*yssd + (1.0f-gg)*yr[i]) * silu_(zz);
}

__global__ void k_ew_add(const float* __restrict__ a, const float* __restrict__ b,
                         float* __restrict__ o, long n){
  long i = (long)blockIdx.x*blockDim.x + threadIdx.x;
  if (i < n) o[i] = a[i] + b[i];
}

__global__ void k_mean_l(const float* __restrict__ in, float* __restrict__ out, int nBD){
  int idx = blockIdx.x*blockDim.x + threadIdx.x;
  if (idx >= nBD) return;
  int b = idx / DD, d = idx % DD;
  float s = 0.f;
  const float* p = in + (size_t)b*LL*DD + d;
  for (int l = 0; l < LL; ++l) s += p[(size_t)l*DD];
  out[idx] = s / (float)LL;
}

__global__ void __launch_bounds__(256)
k_rag(const float* __restrict__ hm, const float* __restrict__ ragst,
      const float* __restrict__ Wq, const float* __restrict__ Wo,
      float* __restrict__ dr){
  const int b = blockIdx.x; const int tid = threadIdx.x;
  __shared__ float q[NNR], info[NNR];
  if (tid < NNR){
    float s = 0.f;
    for (int d = 0; d < DD; ++d) s += hm[(size_t)b*DD + d]*Wq[(size_t)d*NNR + tid];
    q[tid] = s;
  }
  __syncthreads();
  if (tid < NNR){
    float s = 0.f;
    for (int m = 0; m < NNR; ++m) s += ragst[((size_t)b*NNR + tid)*NNR + m]*q[m];
    info[tid] = s;
  }
  __syncthreads();
  for (int d = tid; d < DD; d += 256){
    float s = 0.f;
    for (int n = 0; n < NNR; ++n) s += info[n]*Wo[(size_t)n*DD + d];
    dr[(size_t)b*DD + d] = 0.1f*s;
  }
}

__global__ void k_add_bcast(const float* __restrict__ a, const float* __restrict__ vec,
                            float* __restrict__ o, long n){
  long i = (long)blockIdx.x*blockDim.x + threadIdx.x;
  if (i >= n) return;
  int d = (int)(i % DD); int b = (int)(i / ((long)LL*DD));
  o[i] = a[i] + vec[(size_t)b*DD + d];
}

// Cayley transform: Q = (I-Sk)^-1 (I+Sk), 32x32 Gauss-Jordan in LDS
__global__ void __launch_bounds__(32)
k_cayley(const float* __restrict__ oA, float* __restrict__ Q){
  __shared__ float M[OMD][OMD+1];
  __shared__ float R[OMD][OMD+1];
  const int i = threadIdx.x;
  for (int j = 0; j < OMD; ++j){
    float sk = oA[i*OMD + j] - oA[j*OMD + i];
    M[i][j] = (i == j ? 1.f : 0.f) - sk;
    R[i][j] = (i == j ? 1.f : 0.f) + sk;
  }
  __syncthreads();
  for (int k = 0; k < OMD; ++k){
    if (i == k){
      float piv = 1.f / M[k][k];
      for (int j = 0; j < OMD; ++j){ M[k][j] *= piv; R[k][j] *= piv; }
    }
    __syncthreads();
    if (i != k){
      float f = M[i][k];
      for (int j = 0; j < OMD; ++j){ M[i][j] -= f*M[k][j]; R[i][j] -= f*R[k][j]; }
    }
    __syncthreads();
  }
  for (int j = 0; j < OMD; ++j) Q[i*OMD + j] = R[i][j];
}

__global__ void __launch_bounds__(32)
k_omega_scan(const float* __restrict__ Q, const float* __restrict__ uo,
             float* __restrict__ hs){
  const int b = blockIdx.x; const int i = threadIdx.x;
  __shared__ float Qs[OMD*OMD];
  __shared__ float h[OMD];
  for (int idx = i; idx < OMD*OMD; idx += 32) Qs[idx] = Q[idx];
  h[i] = 0.f;
  __syncthreads();
  for (int t = 0; t < LL; ++t){
    const size_t row = (size_t)b*LL + t;
    float s = uo[row*OMD + i];
#pragma unroll
    for (int j = 0; j < OMD; ++j) s += Qs[i*OMD + j]*h[j];
    __syncthreads();
    h[i] = s;
    hs[row*OMD + i] = s;
    __syncthreads();
  }
}

// MoLE top-2: one block per token
__global__ void __launch_bounds__(256)
k_mole(const float* __restrict__ x4, const float* __restrict__ router,
       const float* __restrict__ mdn, const float* __restrict__ mup,
       float* __restrict__ x5){
  const int t = blockIdx.x; const int tid = threadIdx.x;
  __shared__ float red[256];
  __shared__ float gate[8], down[64], xs[DD];
  const float* xr = x4 + (size_t)t*DD;
  for (int d = tid; d < DD; d += 256) xs[d] = xr[d];
  __syncthreads();
  __shared__ float logits[8];
  for (int e = 0; e < 8; ++e){
    float s = 0.f;
    for (int d = tid; d < DD; d += 256) s += xs[d]*router[(size_t)d*8 + e];
    float tot = blk_reduce(s, red);
    if (tid == 0) logits[e] = tot;
  }
  __syncthreads();
  if (tid == 0){
    float mx = logits[0];
    for (int e = 1; e < 8; ++e) mx = fmaxf(mx, logits[e]);
    float p[8], se = 0.f;
    for (int e = 0; e < 8; ++e){ p[e] = expf(logits[e]-mx); se += p[e]; }
    for (int e = 0; e < 8; ++e) p[e] /= se;
    int e1 = 0;
    for (int e = 1; e < 8; ++e) if (p[e] > p[e1]) e1 = e;
    int e2 = (e1 == 0) ? 1 : 0;
    for (int e = 0; e < 8; ++e) if (e != e1 && p[e] > p[e2]) e2 = e;
    float s2 = p[e1] + p[e2];
    for (int e = 0; e < 8; ++e) gate[e] = 0.f;
    gate[e1] = p[e1]/s2; gate[e2] = p[e2]/s2;
  }
  __syncthreads();
  {
    const int pair = tid & 63, qd = tid >> 6;
    const int e = pair >> 3, r = pair & 7;
    float s = 0.f;
    for (int d = qd*192; d < qd*192 + 192; ++d)
      s += xs[d]*mdn[((size_t)e*DD + d)*8 + r];
    red[tid] = s; __syncthreads();
    if (qd == 0)
      down[pair] = gate[e]*(red[pair] + red[64+pair] + red[128+pair] + red[192+pair]);
    __syncthreads();
  }
  for (int d = tid; d < DD; d += 256){
    float acc = 0.f;
#pragma unroll
    for (int pr = 0; pr < 64; ++pr){
      int e = pr >> 3, r = pr & 7;
      acc += down[pr]*mup[((size_t)e*8 + r)*DD + d];
    }
    x5[(size_t)t*DD + d] = xs[d] + acc;
  }
}

__global__ void __launch_bounds__(256)
k_novelty(const float* __restrict__ hmo, const float* __restrict__ hmn,
          const float* __restrict__ Wn, const float* __restrict__ bn,
          float* __restrict__ nov){
  __shared__ float red[256];
  const int b = blockIdx.x;
  float s = 0.f;
  for (int d = threadIdx.x; d < DD; d += 256)
    s += hmo[(size_t)b*DD + d]*Wn[d] + hmn[(size_t)b*DD + d]*Wn[DD + d];
  float tot = blk_reduce(s, red);
  if (threadIdx.x == 0) nov[b] = sig_(tot + bn[0]);
}

__global__ void k_novmix(const float* __restrict__ x5, const float* __restrict__ xo,
                         const float* __restrict__ nov, float* __restrict__ x6, long n){
  long i = (long)blockIdx.x*blockDim.x + threadIdx.x;
  if (i >= n) return;
  int b = (int)(i / ((long)LL*DD));
  float nv = nov[b];
  x6[i] = nv*x5[i] + (1.0f-nv)*xo[i];
}

__global__ void __launch_bounds__(256)
k_memfuse(const float* __restrict__ hm6, const float* __restrict__ memv,
          const float* __restrict__ Wfp, const float* __restrict__ Wg,
          const float* __restrict__ bg, float* __restrict__ md){
  const int b = blockIdx.x; const int tid = threadIdx.x;
  __shared__ float comb[DD + MEMD];
  __shared__ float red[256];
  __shared__ float mgs;
  for (int i = tid; i < DD + MEMD; i += 256)
    comb[i] = (i < DD) ? hm6[(size_t)b*DD + i] : memv[(size_t)b*MEMD + (i-DD)];
  __syncthreads();
  float s = 0.f;
  for (int i = tid; i < DD + MEMD; i += 256) s += comb[i]*Wg[i];
  float tot = blk_reduce(s, red);
  if (tid == 0) mgs = sig_(tot + bg[0]);
  __syncthreads();
  for (int d = tid; d < DD; d += 256){
    float a = 0.f;
    for (int i = 0; i < DD + MEMD; ++i) a += comb[i]*Wfp[(size_t)i*DD + d];
    md[(size_t)b*DD + d] = mgs*a;
  }
}

// ---------------------------------------------------------------------------
// Host orchestration
// ---------------------------------------------------------------------------
static inline unsigned cdiv_(long a, long b){ return (unsigned)((a + b - 1)/b); }

extern "C" void kernel_launch(void* const* d_in, const int* in_sizes, int n_in,
                              void* d_out, int out_size, void* d_ws, size_t ws_size,
                              hipStream_t stream){
  (void)in_sizes; (void)n_in; (void)out_size; (void)ws_size;
  const float* x       = (const float*)d_in[0];
  const float* wkv0    = (const float*)d_in[1];
  const float* x_prev  = (const float*)d_in[2];
  const float* memv    = (const float*)d_in[3];
  const float* ragst   = (const float*)d_in[4];
  const float* ssd0    = (const float*)d_in[5];
  const float* convst  = (const float*)d_in[6];
  const float* xpl     = (const float*)d_in[7];
  const float* ln_g    = (const float*)d_in[8];
  const float* ln_b    = (const float*)d_in[9];
  const float* W_pred  = (const float*)d_in[10];
  const float* mu      = (const float*)d_in[11];
  const float* W_in    = (const float*)d_in[12];
  const float* conv_w  = (const float*)d_in[13];
  const float* conv_b  = (const float*)d_in[14];
  const float* W_dt    = (const float*)d_in[15];
  const float* dt_bias = (const float*)d_in[16];
  const float* A_log   = (const float*)d_in[17];
  const float* D_skip  = (const float*)d_in[18];
  const float* W_B     = (const float*)d_in[19];
  const float* W_C     = (const float*)d_in[20];
  const float* Wr      = (const float*)d_in[21];
  const float* Wk      = (const float*)d_in[22];
  const float* Wv      = (const float*)d_in[23];
  const float* Ww      = (const float*)d_in[24];
  const float* W_rwkv  = (const float*)d_in[25];
  const float* W_fuse  = (const float*)d_in[26];
  const float* W_out   = (const float*)d_in[27];
  const float* omega_A = (const float*)d_in[28];
  const float* om_down = (const float*)d_in[29];
  const float* om_up   = (const float*)d_in[30];
  const float* router  = (const float*)d_in[31];
  const float* mole_dn = (const float*)d_in[32];
  const float* mole_upw= (const float*)d_in[33];
  const float* W_nov   = (const float*)d_in[34];
  const float* b_nov   = (const float*)d_in[35];
  const float* W_ragq  = (const float*)d_in[36];
  const float* W_rago  = (const float*)d_in[37];
  const float* W_memfp = (const float*)d_in[38];
  const float* W_memg  = (const float*)d_in[39];
  const float* b_memg  = (const float*)d_in[40];
  float* out = (float*)d_out;

  size_t off = 0;
  auto alloc = [&](size_t elems, size_t esz)->void*{
    void* p = (char*)d_ws + off;
    off += (elems*esz + 255) & ~(size_t)255;
    return p;
  };
  unsigned short* wpred_t = (unsigned short*)alloc((size_t)DD*DD, 2);
  unsigned short* win_t   = (unsigned short*)alloc((size_t)D2*DD, 2);
  unsigned short* wdt_t   = (unsigned short*)alloc((size_t)16*DD, 2);
  unsigned short* wb_t    = (unsigned short*)alloc((size_t)NNR*DD, 2);
  unsigned short* wc_t    = (unsigned short*)alloc((size_t)NNR*DD, 2);
  unsigned short* wrr_t   = (unsigned short*)alloc((size_t)NNR*DD, 2);
  unsigned short* wk_t    = (unsigned short*)alloc((size_t)NNR*DD, 2);
  unsigned short* wv_t    = (unsigned short*)alloc((size_t)NNR*DD, 2);
  unsigned short* ww_t    = (unsigned short*)alloc((size_t)NNR*DD, 2);
  unsigned short* wrwkv_t = (unsigned short*)alloc((size_t)DD*NNR, 2);
  unsigned short* wout_t  = (unsigned short*)alloc((size_t)DD*DD, 2);
  unsigned short* wod_t   = (unsigned short*)alloc((size_t)OMD*DD, 2);
  unsigned short* wou_t   = (unsigned short*)alloc((size_t)DD*OMD, 2);
  unsigned short* abuf = (unsigned short*)alloc((size_t)TT*DD, 2);
  unsigned short* sbuf = (unsigned short*)alloc((size_t)TT*NNR, 2);
  float* f1   = (float*)alloc((size_t)TT*DD, 4);   // pred -> yr -> core -> odelta
  float* bufA = (float*)alloc((size_t)TT*DD, 4);   // xn -> xc -> x3
  float* bufB = (float*)alloc((size_t)TT*DD, 4);   // x1 -> ys -> x2 -> x4 -> x6
  float* bufC = (float*)alloc((size_t)TT*DD, 4);   // xm -> cp -> x5
  float* xz   = (float*)alloc((size_t)TT*D2, 4);
  float* dtp  = (float*)alloc((size_t)TT*HH, 4);
  float* dtb  = (float*)alloc((size_t)TT*HH, 4);
  float* dab  = (float*)alloc((size_t)TT*HH, 4);
  float* Bm   = (float*)alloc((size_t)TT*NNR, 4);
  float* Cm   = (float*)alloc((size_t)TT*NNR, 4);
  float* rrb  = (float*)alloc((size_t)TT*NNR, 4);
  float* kkb  = (float*)alloc((size_t)TT*NNR, 4);
  float* vvb  = (float*)alloc((size_t)TT*NNR, 4);
  float* wwb  = (float*)alloc((size_t)TT*NNR, 4);
  float* osb  = (float*)alloc((size_t)TT*NNR, 4);
  float* gb   = (float*)alloc((size_t)TT, 4);
  float* uob  = (float*)alloc((size_t)TT*OMD, 4);
  float* hsb  = (float*)alloc((size_t)TT*OMD, 4);
  float* Qb   = (float*)alloc((size_t)OMD*OMD, 4);
  float* hm2  = (float*)alloc((size_t)BB*DD, 4);
  float* hmo  = (float*)alloc((size_t)BB*DD, 4);
  float* hmn  = (float*)alloc((size_t)BB*DD, 4);
  float* hm6  = (float*)alloc((size_t)BB*DD, 4);
  float* drb  = (float*)alloc((size_t)BB*DD, 4);
  float* mdb  = (float*)alloc((size_t)BB*DD, 4);
  float* novb = (float*)alloc((size_t)BB, 4);

  const long nTD = (long)TT*DD;
  const dim3 blk256(256);

  auto wconv = [&](const float* src, unsigned short* dst, int K, int Nout, int Npad){
    k_wconv_t<<<cdiv_((long)Npad*K,256), blk256, 0, stream>>>(src, dst, K, Nout, Npad);
  };
  auto gemm = [&](const unsigned short* A, const unsigned short* Wt, float* C,
                  int K, int Nout){
    if ((K % 64 == 0) && (Nout % 64 == 0)){
      dim3 grid(TT/64, (unsigned)(Nout/64));
      k_gemm_wmma_tdm<<<grid, dim3(128), 0, stream>>>((const __bf16*)A, (const __bf16*)Wt, C, K, Nout);
    } else {
      dim3 grid(TT/16, cdiv_(Nout,16));
      k_gemm_wmma<<<grid, dim3(32), 0, stream>>>((const __bf16*)A, (const __bf16*)Wt, C, K, Nout);
    }
  };
  auto cvt = [&](const float* src, unsigned short* dst, long n){
    k_cvt_bf16<<<cdiv_(n,256), blk256, 0, stream>>>(src, dst, n);
  };

  // --- weight conversion (f32 -> bf16, transposed, padded) ---
  wconv(W_pred, wpred_t, DD, DD, DD);
  wconv(W_in,   win_t,   DD, D2, D2);
  wconv(W_dt,   wdt_t,   DD, HH, 16);
  wconv(W_B,    wb_t,    DD, NNR, NNR);
  wconv(W_C,    wc_t,    DD, NNR, NNR);
  wconv(Wr,     wrr_t,   DD, NNR, NNR);
  wconv(Wk,     wk_t,    DD, NNR, NNR);
  wconv(Wv,     wv_t,    DD, NNR, NNR);
  wconv(Ww,     ww_t,    DD, NNR, NNR);
  wconv(W_rwkv, wrwkv_t, NNR, DD, DD);
  wconv(W_out,  wout_t,  DD, DD, DD);
  wconv(om_down,wod_t,   DD, OMD, OMD);
  wconv(om_up,  wou_t,   OMD, DD, DD);

  // --- predictive residual: x1 = x + 0.05*(x - xpl @ W_pred) ---
  cvt(xpl, abuf, nTD);
  gemm(abuf, wpred_t, f1, DD, DD);
  k_ew_pred<<<cdiv_(nTD,256), blk256, 0, stream>>>(x, f1, bufB, nTD);

  // --- LN + token shift ---
  k_ln<<<TT, blk256, 0, stream>>>(bufB, ln_g, ln_b, bufA);
  k_shift_mix<<<cdiv_(nTD,256), blk256, 0, stream>>>(bufA, x_prev, mu, bufC, nTD);

  // --- xm projections ---
  cvt(bufC, abuf, nTD);
  gemm(abuf, win_t, xz, DD, D2);
  k_conv_silu<<<cdiv_(nTD,256), blk256, 0, stream>>>(xz, convst, conv_w, conv_b, bufA, nTD);
  gemm(abuf, wdt_t, dtp, DD, HH);
  k_ew_dt<<<cdiv_((long)TT*HH,256), blk256, 0, stream>>>(dtp, dt_bias, A_log, dtb, dab, (long)TT*HH);
  gemm(abuf, wb_t,  Bm,  DD, NNR);
  gemm(abuf, wc_t,  Cm,  DD, NNR);
  gemm(abuf, wrr_t, rrb, DD, NNR);
  gemm(abuf, wk_t,  kkb, DD, NNR);
  gemm(abuf, wv_t,  vvb, DD, NNR);
  gemm(abuf, ww_t,  wwb, DD, NNR);
  k_ew_sigmoid  <<<cdiv_((long)TT*NNR,256), blk256, 0, stream>>>(rrb, (long)TT*NNR);
  k_ew_expnegexp<<<cdiv_((long)TT*NNR,256), blk256, 0, stream>>>(wwb, (long)TT*NNR);

  // --- sequential scans ---
  k_ssd_scan <<<BB*HH, blk256, 0, stream>>>(dab, dtb, bufA, Bm, Cm, ssd0, bufB);
  k_rwkv_scan<<<BB,    blk256, 0, stream>>>(wwb, kkb, vvb, rrb, wkv0, osb);
  k_gfuse<<<cdiv_(TT,256), blk256, 0, stream>>>(bufC, W_fuse, gb, TT);

  // --- fusion + output projection ---
  cvt(osb, sbuf, (long)TT*NNR);
  gemm(sbuf, wrwkv_t, f1, NNR, DD);                 // y_rwkv
  k_combine<<<cdiv_(nTD,256), blk256, 0, stream>>>(gb, bufB, bufA, f1, xz, D_skip, bufC, nTD);
  cvt(bufC, abuf, nTD);
  gemm(abuf, wout_t, f1, DD, DD);                   // core
  k_ew_add<<<cdiv_(nTD,256), blk256, 0, stream>>>(x, f1, bufB, nTD);   // x2

  // --- RAG ---
  k_mean_l<<<cdiv_((long)BB*DD,256), blk256, 0, stream>>>(bufB, hm2, BB*DD);
  k_rag<<<BB, blk256, 0, stream>>>(hm2, ragst, W_ragq, W_rago, drb);
  k_add_bcast<<<cdiv_(nTD,256), blk256, 0, stream>>>(bufB, drb, bufA, nTD); // x3

  // --- orthogonal (Cayley) RNN ---
  k_cayley<<<1, dim3(32), 0, stream>>>(omega_A, Qb);
  cvt(bufA, abuf, nTD);
  gemm(abuf, wod_t, uob, DD, OMD);
  k_omega_scan<<<BB, dim3(32), 0, stream>>>(Qb, uob, hsb);
  cvt(hsb, sbuf, (long)TT*OMD);
  gemm(sbuf, wou_t, f1, OMD, DD);                   // odelta
  k_ew_add<<<cdiv_(nTD,256), blk256, 0, stream>>>(bufA, f1, bufB, nTD);  // x4

  // --- MoLE ---
  k_mole<<<TT, blk256, 0, stream>>>(bufB, router, mole_dn, mole_upw, bufC); // x5

  // --- novelty gate ---
  k_mean_l<<<cdiv_((long)BB*DD,256), blk256, 0, stream>>>(x,    hmo, BB*DD);
  k_mean_l<<<cdiv_((long)BB*DD,256), blk256, 0, stream>>>(bufC, hmn, BB*DD);
  k_novelty<<<BB, blk256, 0, stream>>>(hmo, hmn, W_nov, b_nov, novb);
  k_novmix<<<cdiv_(nTD,256), blk256, 0, stream>>>(bufC, x, novb, bufB, nTD); // x6

  // --- memory fusion + output ---
  k_mean_l<<<cdiv_((long)BB*DD,256), blk256, 0, stream>>>(bufB, hm6, BB*DD);
  k_memfuse<<<BB, blk256, 0, stream>>>(hm6, memv, W_memfp, W_memg, b_memg, mdb);
  k_add_bcast<<<cdiv_(nTD,256), blk256, 0, stream>>>(bufB, mdb, out, nTD);
}
